// LinearAttention2D_2001454760423
// MI455X (gfx1250) — compile-verified, matrix operand order checked by reading, not run on hardware
//
#include <hip/hip_runtime.h>
#include <hip/hip_bf16.h>
#include <math.h>

// ---------------------------------------------------------------------------
// Types for CDNA5 WMMA (gfx1250, wave32)
// ---------------------------------------------------------------------------
typedef __attribute__((ext_vector_type(16))) __bf16 v16bf;
typedef __attribute__((ext_vector_type(8)))  __bf16 v8bf;
typedef __attribute__((ext_vector_type(8)))  float  v8f;

#define WAVE 32

// Problem constants
#define BATCH   8
#define DIM     256
#define HEADS   8
#define DHEAD   64
#define HIDDEN  512          // HEADS * DHEAD
#define NPOS    4096         // 64*64
#define NMEM    8
#define NKV     4104         // NPOS + NMEM
#define NKVP    4128         // padded to multiple of 32
#define OQKV    1536         // 3*HIDDEN

// ---------------------------------------------------------------------------
// Fragment loader: 16-bit A/B operand, 16 rows (or cols) x 32 K.
// Memory layout assumed: row-major [16 rows][ld], K contiguous within a row.
// Per ISA 7.12.2: lane = row (mod 16); lanes 0-15 take K-halves {0..7,16..23},
// lanes 16-31 take {8..15,24..31}. Two 16B loads per lane.
// ---------------------------------------------------------------------------
__device__ __forceinline__ v16bf load_frag16(const __bf16* __restrict__ base,
                                             int row0, int ld, int k0, int lane) {
  const int r  = row0 + (lane & 15);
  const int hi = (lane >> 4) & 1;
  const __bf16* p = base + (size_t)r * ld + k0 + hi * 8;
  v8bf c0 = *(const v8bf*)(p);
  v8bf c1 = *(const v8bf*)(p + 16);
  v16bf out;
#pragma unroll
  for (int i = 0; i < 8; ++i) { out[i] = c0[i]; out[i + 8] = c1[i]; }
  return out;
}

__device__ __forceinline__ v8f wmma_bf16(v16bf a, v16bf b, v8f c) {
  return __builtin_amdgcn_wmma_f32_16x16x32_bf16(
      /*neg_a=*/false, a, /*neg_b=*/false, b,
      /*c_mod=*/(short)0, c, /*reuse_a=*/false, /*reuse_b=*/false);
}

// ---------------------------------------------------------------------------
// 0) fp32 -> bf16 weight conversion
// ---------------------------------------------------------------------------
__global__ void k_cvt_bf16(const float* __restrict__ src, __bf16* __restrict__ dst, int n) {
  int i = blockIdx.x * blockDim.x + threadIdx.x;
  if (i < n) dst[i] = (__bf16)src[i];
}

// ---------------------------------------------------------------------------
// 1) RMS norm over channels + transpose to [b][pos][c] bf16 (WMMA B layout)
//    x: [b][256][4096] fp32
// ---------------------------------------------------------------------------
__global__ void k_rms1(const float* __restrict__ x, const float* __restrict__ nw,
                       __bf16* __restrict__ xn) {
  const int b = blockIdx.y;
  const int p = blockIdx.x * blockDim.x + threadIdx.x;   // 0..4095
  const float* xb = x + (size_t)b * DIM * NPOS;
  float ss = 0.f;
#pragma unroll 4
  for (int c = 0; c < DIM; ++c) {
    float v = xb[(size_t)c * NPOS + p];
    ss += v * v;
  }
  float nrm = fmaxf(sqrtf(ss), 1e-12f);
  float inv = 16.0f / nrm;                                // sqrt(256) / ||x||
  __bf16* o = xn + ((size_t)b * NPOS + p) * DIM;
#pragma unroll 4
  for (int c = 0; c < DIM; ++c)
    o[c] = (__bf16)(xb[(size_t)c * NPOS + p] * inv * nw[c]);
}

// ---------------------------------------------------------------------------
// 2) QKV GEMM: [1536 x 4096] = w_qkv[1536 x 256] * xn^T, per batch.
//    Register-blocked: each wave computes 32(M) x 64(N) = 2x4 WMMA tiles;
//    A fragments reused across 4 N-tiles, B across 2 M-tiles.
//    Block 256 thr = 8 waves (2 along M x 4 along N) -> 64 x 256 per block.
//    grid (16, 24, 8).
//    Scatters q -> QS[b][h][n][64], k/v -> KS/VS[b][h][d][4128] at col 8+n.
// ---------------------------------------------------------------------------
__global__ void __launch_bounds__(256)
k_gemm_qkv(const __bf16* __restrict__ wq, const __bf16* __restrict__ xn,
           __bf16* __restrict__ QS, __bf16* __restrict__ KS, __bf16* __restrict__ VS) {
  const int lane = threadIdx.x & (WAVE - 1);
  const int wv   = threadIdx.x / WAVE;
  const int b    = blockIdx.z;
  const int obase = blockIdx.y * 64 + (wv >> 2) * 32;   // 32 rows per wave
  const int pbase = blockIdx.x * 256 + (wv & 3) * 64;   // 64 cols per wave

  const __bf16* xb = xn + (size_t)b * NPOS * DIM;       // rows = positions

  v8f acc[2][4] = {};
#pragma unroll
  for (int kk = 0; kk < DIM; kk += 32) {
    v16bf a0 = load_frag16(wq, obase,      DIM, kk, lane);
    v16bf a1 = load_frag16(wq, obase + 16, DIM, kk, lane);
    v16bf bf0 = load_frag16(xb, pbase,      DIM, kk, lane);
    v16bf bf1 = load_frag16(xb, pbase + 16, DIM, kk, lane);
    v16bf bf2 = load_frag16(xb, pbase + 32, DIM, kk, lane);
    v16bf bf3 = load_frag16(xb, pbase + 48, DIM, kk, lane);
    acc[0][0] = wmma_bf16(a0, bf0, acc[0][0]);
    acc[0][1] = wmma_bf16(a0, bf1, acc[0][1]);
    acc[0][2] = wmma_bf16(a0, bf2, acc[0][2]);
    acc[0][3] = wmma_bf16(a0, bf3, acc[0][3]);
    acc[1][0] = wmma_bf16(a1, bf0, acc[1][0]);
    acc[1][1] = wmma_bf16(a1, bf1, acc[1][1]);
    acc[1][2] = wmma_bf16(a1, bf2, acc[1][2]);
    acc[1][3] = wmma_bf16(a1, bf3, acc[1][3]);
  }

  const int hi = (lane >> 4) & 1;
#pragma unroll
  for (int ti = 0; ti < 2; ++ti) {
    const int obt = obase + 16 * ti;
    const int seg  = obt >> 9;                          // 0=q 1=k 2=v
    const int o512 = obt & 511;
    const int h = o512 >> 6;
    const int dbase = (o512 & 63) + 8 * hi;             // 8 consecutive d values
#pragma unroll
    for (int tj = 0; tj < 4; ++tj) {
      const int n = pbase + 16 * tj + (lane & 15);
      if (seg == 0) {
        __bf16* dst = QS + (((size_t)(b * HEADS + h) * NPOS + n) * DHEAD) + dbase;
#pragma unroll
        for (int i = 0; i < 8; ++i) dst[i] = (__bf16)acc[ti][tj][i];
      } else {
        __bf16* base = (seg == 1 ? KS : VS);
#pragma unroll
        for (int i = 0; i < 8; ++i) {
          size_t row = (size_t)(b * HEADS + h) * DHEAD + (dbase + i);
          base[row * NKVP + NMEM + n] = (__bf16)acc[ti][tj][i];
        }
      }
    }
  }
}

// ---------------------------------------------------------------------------
// 3) Fill mem_kv columns (0..7) and zero tail padding (4104..4127).
//    mem_kv: [2][8][64][8] fp32. One thread per (b,h,d) row.
// ---------------------------------------------------------------------------
__global__ void k_memfill(const float* __restrict__ mem_kv,
                          __bf16* __restrict__ KS, __bf16* __restrict__ VS) {
  int idx = blockIdx.x * blockDim.x + threadIdx.x;    // 0..4095 = (b*8+h)*64+d
  int d = idx & 63;
  int h = (idx >> 6) & 7;
  __bf16* kr = KS + (size_t)idx * NKVP;
  __bf16* vr = VS + (size_t)idx * NKVP;
#pragma unroll
  for (int j = 0; j < NMEM; ++j) {
    kr[j] = (__bf16)mem_kv[((0 * HEADS + h) * DHEAD + d) * NMEM + j];
    vr[j] = (__bf16)mem_kv[((1 * HEADS + h) * DHEAD + d) * NMEM + j];
  }
  for (int j = NKV; j < NKVP; ++j) { kr[j] = (__bf16)0.f; vr[j] = (__bf16)0.f; }
}

// ---------------------------------------------------------------------------
// 4a) softmax over n (4104) for k rows. One 256-thread block per (b,h,d).
// ---------------------------------------------------------------------------
__global__ void __launch_bounds__(256)
k_softmax_k(__bf16* __restrict__ KS) {
  __shared__ float red[256];
  const int t = threadIdx.x;
  __bf16* kr = KS + (size_t)blockIdx.x * NKVP;

  float mx = -1e30f;
  for (int j = t; j < NKV; j += 256) mx = fmaxf(mx, (float)kr[j]);
  red[t] = mx; __syncthreads();
  for (int s = 128; s > 0; s >>= 1) { if (t < s) red[t] = fmaxf(red[t], red[t + s]); __syncthreads(); }
  mx = red[0]; __syncthreads();

  float sum = 0.f;
  for (int j = t; j < NKV; j += 256) sum += __expf((float)kr[j] - mx);
  red[t] = sum; __syncthreads();
  for (int s = 128; s > 0; s >>= 1) { if (t < s) red[t] += red[t + s]; __syncthreads(); }
  float inv = 1.0f / red[0];

  for (int j = t; j < NKV; j += 256) kr[j] = (__bf16)(__expf((float)kr[j] - mx) * inv);
}

// ---------------------------------------------------------------------------
// 4b) softmax over d (64) for q rows, fused * dh^-0.5. One thread per (b,h,n).
// ---------------------------------------------------------------------------
__global__ void k_softmax_q(__bf16* __restrict__ QS) {
  size_t row = (size_t)blockIdx.x * blockDim.x + threadIdx.x;  // 262144 rows
  __bf16* q = QS + row * DHEAD;
  float mx = -1e30f;
#pragma unroll 4
  for (int i = 0; i < DHEAD; ++i) mx = fmaxf(mx, (float)q[i]);
  float s = 0.f;
#pragma unroll 4
  for (int i = 0; i < DHEAD; ++i) s += __expf((float)q[i] - mx);
  float inv = 0.125f / s;                                      // 1/sqrt(64) / sum
#pragma unroll 4
  for (int i = 0; i < DHEAD; ++i) q[i] = (__bf16)(__expf((float)q[i] - mx) * inv);
}

// ---------------------------------------------------------------------------
// 5) context[d][e] = sum_n k[d][n] * v[e][n], K = 4128 (padded).
//    Stored transposed: CT[b][h][e][64] bf16 (A operand for next GEMM).
//    One block (512 thr = 16 waves) per (b,h); wave -> one 16x16 tile.
// ---------------------------------------------------------------------------
__global__ void __launch_bounds__(512)
k_gemm_ctx(const __bf16* __restrict__ KS, const __bf16* __restrict__ VS,
           __bf16* __restrict__ CT) {
  const int lane = threadIdx.x & (WAVE - 1);
  const int wv   = threadIdx.x / WAVE;
  const int bh   = blockIdx.x;
  const int dt   = (wv >> 2) * 16;
  const int et   = (wv & 3) * 16;

  const __bf16* ka = KS + (size_t)bh * DHEAD * NKVP;
  const __bf16* vb = VS + (size_t)bh * DHEAD * NKVP;

  v8f acc = {};
  for (int kk = 0; kk < NKVP; kk += 32) {
    v16bf a = load_frag16(ka, dt, NKVP, kk, lane);
    v16bf bfr = load_frag16(vb, et, NKVP, kk, lane);
    acc = wmma_bf16(a, bfr, acc);
  }
  // D: lane&15 = e column, element i -> d row. Write transposed (row e, col d).
  const int e  = et + (lane & 15);
  const int d0 = dt + 8 * ((lane >> 4) & 1);
  __bf16* dst = CT + ((size_t)bh * DHEAD + e) * DHEAD + d0;
#pragma unroll
  for (int i = 0; i < 8; ++i) dst[i] = (__bf16)acc[i];
}

// ---------------------------------------------------------------------------
// 6) out[e][n] = sum_d ctxT[e][d] * q[d][n], K = 64, per (b,h).
//    Wave computes 16(e) x 64(n): A fragment reused across 4 N-tiles.
//    Block 256 thr = 8 waves (4 e-tiles x 2 n-groups) -> 64 x 128 per block.
//    grid (32, 64). Scatter to ATTN[b][n][512] at o = h*64+e.
// ---------------------------------------------------------------------------
__global__ void __launch_bounds__(256)
k_gemm_attn(const __bf16* __restrict__ CT, const __bf16* __restrict__ QS,
            __bf16* __restrict__ ATTN) {
  const int lane = threadIdx.x & (WAVE - 1);
  const int wv   = threadIdx.x / WAVE;
  const int bh   = blockIdx.y;
  const int b    = bh >> 3;
  const int h    = bh & 7;
  const int et   = (wv >> 1) * 16;
  const int nt   = blockIdx.x * 128 + (wv & 1) * 64;

  const __bf16* ca = CT + (size_t)bh * DHEAD * DHEAD;    // rows e, ld 64
  const __bf16* qb = QS + (size_t)bh * NPOS * DHEAD;     // rows n, ld 64

  v8f acc[4] = {};
#pragma unroll
  for (int kk = 0; kk < DHEAD; kk += 32) {
    v16bf a = load_frag16(ca, et, DHEAD, kk, lane);
#pragma unroll
    for (int tj = 0; tj < 4; ++tj) {
      v16bf bfr = load_frag16(qb, nt + 16 * tj, DHEAD, kk, lane);
      acc[tj] = wmma_bf16(a, bfr, acc[tj]);
    }
  }
  const int e0 = et + 8 * ((lane >> 4) & 1);
#pragma unroll
  for (int tj = 0; tj < 4; ++tj) {
    const int n = nt + 16 * tj + (lane & 15);
    __bf16* dst = ATTN + ((size_t)b * NPOS + n) * HIDDEN + h * DHEAD + e0;
#pragma unroll
    for (int i = 0; i < 8; ++i) dst[i] = (__bf16)acc[tj][i];
  }
}

// ---------------------------------------------------------------------------
// 7) y[c][p] = sum_o w_out[c][o] * attn[p][o] + b_out[c]; fp32 out [b][256][4096].
//    Register-blocked 32 x 64 per wave like the qkv GEMM. grid (16, 4, 8).
// ---------------------------------------------------------------------------
__global__ void __launch_bounds__(256)
k_gemm_y(const __bf16* __restrict__ wo, const __bf16* __restrict__ ATTN,
         const float* __restrict__ bias, float* __restrict__ Y) {
  const int lane = threadIdx.x & (WAVE - 1);
  const int wv   = threadIdx.x / WAVE;
  const int b    = blockIdx.z;
  const int cbase = blockIdx.y * 64 + (wv >> 2) * 32;
  const int pbase = blockIdx.x * 256 + (wv & 3) * 64;

  const __bf16* ab = ATTN + (size_t)b * NPOS * HIDDEN;   // rows p, ld 512

  v8f acc[2][4] = {};
#pragma unroll
  for (int kk = 0; kk < HIDDEN; kk += 32) {
    v16bf a0 = load_frag16(wo, cbase,      HIDDEN, kk, lane);
    v16bf a1 = load_frag16(wo, cbase + 16, HIDDEN, kk, lane);
    v16bf bf0 = load_frag16(ab, pbase,      HIDDEN, kk, lane);
    v16bf bf1 = load_frag16(ab, pbase + 16, HIDDEN, kk, lane);
    v16bf bf2 = load_frag16(ab, pbase + 32, HIDDEN, kk, lane);
    v16bf bf3 = load_frag16(ab, pbase + 48, HIDDEN, kk, lane);
    acc[0][0] = wmma_bf16(a0, bf0, acc[0][0]);
    acc[0][1] = wmma_bf16(a0, bf1, acc[0][1]);
    acc[0][2] = wmma_bf16(a0, bf2, acc[0][2]);
    acc[0][3] = wmma_bf16(a0, bf3, acc[0][3]);
    acc[1][0] = wmma_bf16(a1, bf0, acc[1][0]);
    acc[1][1] = wmma_bf16(a1, bf1, acc[1][1]);
    acc[1][2] = wmma_bf16(a1, bf2, acc[1][2]);
    acc[1][3] = wmma_bf16(a1, bf3, acc[1][3]);
  }
  const int hi = (lane >> 4) & 1;
#pragma unroll
  for (int ti = 0; ti < 2; ++ti) {
    const int c0 = cbase + 16 * ti + 8 * hi;
#pragma unroll
    for (int tj = 0; tj < 4; ++tj) {
      const int p = pbase + 16 * tj + (lane & 15);
#pragma unroll
      for (int i = 0; i < 8; ++i) {
        int c = c0 + i;
        Y[((size_t)b * DIM + c) * NPOS + p] = acc[ti][tj][i] + bias[c];
      }
    }
  }
}

// ---------------------------------------------------------------------------
// 8) Final RMS norm over channels -> d_out (fp32, [b][256][64][64]).
// ---------------------------------------------------------------------------
__global__ void k_rms2(const float* __restrict__ Y, const float* __restrict__ w,
                       float* __restrict__ out) {
  const int b = blockIdx.y;
  const int p = blockIdx.x * blockDim.x + threadIdx.x;
  const float* yb = Y + (size_t)b * DIM * NPOS;
  float ss = 0.f;
#pragma unroll 4
  for (int c = 0; c < DIM; ++c) {
    float v = yb[(size_t)c * NPOS + p];
    ss += v * v;
  }
  float nrm = fmaxf(sqrtf(ss), 1e-12f);
  float inv = 16.0f / nrm;
  float* ob = out + (size_t)b * DIM * NPOS;
#pragma unroll 4
  for (int c = 0; c < DIM; ++c)
    ob[(size_t)c * NPOS + p] = yb[(size_t)c * NPOS + p] * inv * w[c];
}

// ---------------------------------------------------------------------------
// Host-side launch
// ---------------------------------------------------------------------------
extern "C" void kernel_launch(void* const* d_in, const int* in_sizes, int n_in,
                              void* d_out, int out_size, void* d_ws, size_t ws_size,
                              hipStream_t stream) {
  const float* x      = (const float*)d_in[0];
  const float* norm_w = (const float*)d_in[1];
  const float* w_qkv  = (const float*)d_in[2];
  const float* mem_kv = (const float*)d_in[3];
  const float* w_out  = (const float*)d_in[4];
  const float* b_out  = (const float*)d_in[5];
  const float* onw    = (const float*)d_in[6];
  float* out = (float*)d_out;

  char* ws = (char*)d_ws;
  size_t off = 0;
  auto alloc = [&](size_t bytes) { char* p = ws + off; off += (bytes + 255) & ~(size_t)255; return p; };

  __bf16* WQ = (__bf16*)alloc((size_t)OQKV * DIM * 2);                 // 0.75 MB
  __bf16* WO = (__bf16*)alloc((size_t)DIM * HIDDEN * 2);               // 0.25 MB
  __bf16* XN = (__bf16*)alloc((size_t)BATCH * NPOS * DIM * 2);         // 16.8 MB
  __bf16* QS = (__bf16*)alloc((size_t)BATCH * HEADS * NPOS * DHEAD * 2);   // 33.5 MB
  __bf16* KS = (__bf16*)alloc((size_t)BATCH * HEADS * DHEAD * NKVP * 2);   // 33.8 MB
  __bf16* VS = (__bf16*)alloc((size_t)BATCH * HEADS * DHEAD * NKVP * 2);   // 33.8 MB
  __bf16* CT = (__bf16*)alloc((size_t)BATCH * HEADS * DHEAD * DHEAD * 2);  // 0.5 MB
  __bf16* AT = (__bf16*)alloc((size_t)BATCH * NPOS * HIDDEN * 2);      // 33.5 MB
  float*  Y  = (float*) alloc((size_t)BATCH * DIM * NPOS * 4);         // 33.5 MB
  (void)ws_size; (void)n_in; (void)in_sizes; (void)out_size;

  // 0) weight conversion
  k_cvt_bf16<<<(OQKV * DIM + 255) / 256, 256, 0, stream>>>(w_qkv, WQ, OQKV * DIM);
  k_cvt_bf16<<<(DIM * HIDDEN + 255) / 256, 256, 0, stream>>>(w_out, WO, DIM * HIDDEN);
  // 1) rms norm + transpose to bf16
  k_rms1<<<dim3(NPOS / 256, BATCH), 256, 0, stream>>>(x, norm_w, XN);
  // 2) qkv GEMM (WMMA, 32x64 per wave) with fused scatter
  k_gemm_qkv<<<dim3(NPOS / 256, OQKV / 64, BATCH), 256, 0, stream>>>(WQ, XN, QS, KS, VS);
  // 3) mem_kv columns + padding
  k_memfill<<<(BATCH * HEADS * DHEAD) / 256, 256, 0, stream>>>(mem_kv, KS, VS);
  // 4) softmaxes
  k_softmax_k<<<BATCH * HEADS * DHEAD, 256, 0, stream>>>(KS);
  k_softmax_q<<<(BATCH * HEADS * NPOS) / 256, 256, 0, stream>>>(QS);
  // 5) context GEMM (WMMA)
  k_gemm_ctx<<<BATCH * HEADS, 512, 0, stream>>>(KS, VS, CT);
  // 6) attention output GEMM (WMMA, 16x64 per wave)
  k_gemm_attn<<<dim3(NPOS / 128, BATCH * HEADS), 256, 0, stream>>>(CT, QS, AT);
  // 7) output projection GEMM (WMMA, 32x64 per wave) + bias
  k_gemm_y<<<dim3(NPOS / 256, DIM / 64, BATCH), 256, 0, stream>>>(WO, AT, b_out, Y);
  // 8) final rms norm
  k_rms2<<<dim3(NPOS / 256, BATCH), 256, 0, stream>>>(Y, onw, out);
}
